// GPT_82875688944000
// MI455X (gfx1250) — matrix operand order repaired and checked by reading, not compiled, not run on hardware
//
#include <hip/hip_runtime.h>
#include <hip/hip_bf16.h>

typedef __bf16 v16bf __attribute__((ext_vector_type(16)));
typedef float  v8f   __attribute__((ext_vector_type(8)));

union FragU { unsigned short u[16]; v16bf b; };
struct __align__(16) U4 { unsigned int x, y, z, w; };

__device__ __forceinline__ unsigned short f2bf(float f) {
    unsigned int u = __builtin_bit_cast(unsigned int, f);
    u += 0x7FFFu + ((u >> 16) & 1u);           // round-to-nearest-even
    return (unsigned short)(u >> 16);
}

// ---------------------------------------------------------------------------
// Embedding: x[row] = wte[idx[row]] + wpe[row % T], E = 512, block = 256
// ---------------------------------------------------------------------------
__global__ __launch_bounds__(256) void embed_kernel(const int* __restrict__ idx,
                                                    const float* __restrict__ wte,
                                                    const float* __restrict__ wpe,
                                                    float* __restrict__ X) {
    int row = blockIdx.x, tid = threadIdx.x;
    int t = row & 1023;                         // T = 1024
    int tok = idx[row];
    const float* te = wte + (size_t)tok * 512;
    const float* pe = wpe + (size_t)t * 512;
    float* xr = X + (size_t)row * 512;
    xr[tid]       = te[tid]       + pe[tid];
    xr[tid + 256] = te[tid + 256] + pe[tid + 256];
}

// ---------------------------------------------------------------------------
// LayerNorm over E=512, one block (256 threads) per row
// ---------------------------------------------------------------------------
__global__ __launch_bounds__(256) void ln_kernel(const float* __restrict__ X,
                                                 const float* __restrict__ w,
                                                 const float* __restrict__ bch,
                                                 float* __restrict__ Y) {
    __shared__ float s1[256], s2[256];
    int row = blockIdx.x, tid = threadIdx.x;
    const float* xr = X + (size_t)row * 512;
    float a = xr[tid], b = xr[tid + 256];
    s1[tid] = a + b;
    s2[tid] = a * a + b * b;
    __syncthreads();
    for (int s = 128; s > 0; s >>= 1) {
        if (tid < s) { s1[tid] += s1[tid + s]; s2[tid] += s2[tid + s]; }
        __syncthreads();
    }
    float mu  = s1[0] * (1.0f / 512.0f);
    float var = s2[0] * (1.0f / 512.0f) - mu * mu;
    float r   = rsqrtf(var + 1e-5f);
    float* yr = Y + (size_t)row * 512;
    yr[tid]       = (a - mu) * r * w[tid]       + bch[tid];
    yr[tid + 256] = (b - mu) * r * w[tid + 256] + bch[tid + 256];
}

// ---------------------------------------------------------------------------
// KAN activation pack: out row = [silu(x_0..In-1) | bspl(x_0)*8 | bspl(x_1)*8 ...]
// bf16, row stride In*9.  Uniform cubic B-splines, grid g_t = 0.4 t - 2.2.
// ---------------------------------------------------------------------------
__global__ __launch_bounds__(256) void kan_prep_act(unsigned short* __restrict__ out,
                                                    const float* __restrict__ X,
                                                    int Mrows, int In) {
    size_t gid = (size_t)blockIdx.x * 256 + threadIdx.x;
    size_t total = (size_t)Mrows * In;
    if (gid >= total) return;
    int row = (int)(gid / In);
    int i   = (int)(gid % In);
    float xv = X[gid];
    size_t rb = (size_t)row * (size_t)In * 9;
    float s = xv / (1.0f + __expf(-xv));        // silu
    out[rb + i] = f2bf(s);

    float bb[11];
#pragma unroll
    for (int t = 0; t < 11; ++t) {
        float gl = 0.4f * t - 2.2f;
        bb[t] = (xv >= gl && xv < gl + 0.4f) ? 1.0f : 0.0f;
    }
#pragma unroll
    for (int k = 1; k <= 3; ++k) {
        float inv = 1.0f / (0.4f * (float)k);
#pragma unroll
        for (int t = 0; t + k < 11; ++t) {
            float gl  = 0.4f * t - 2.2f;
            float gr1 = 0.4f * (float)(t + k + 1) - 2.2f;
            bb[t] = (xv - gl) * inv * bb[t] + (gr1 - xv) * inv * bb[t + 1];
        }
    }
    size_t sb = rb + In + (size_t)i * 8;
#pragma unroll
    for (int j = 0; j < 8; ++j) out[sb + j] = f2bf(bb[j]);
}

// ---------------------------------------------------------------------------
// KAN weight pack: Wb[n] = [base_w[n,:] | (spline[n,i,:]*scaler[n,i]) ...] bf16
// ---------------------------------------------------------------------------
__global__ __launch_bounds__(256) void kan_prep_w(unsigned short* __restrict__ out,
                                                  const float* __restrict__ base,
                                                  const float* __restrict__ spline,
                                                  const float* __restrict__ scaler,
                                                  int Out, int In) {
    size_t gid = (size_t)blockIdx.x * 256 + threadIdx.x;
    size_t total = (size_t)Out * In;
    if (gid >= total) return;
    int n = (int)(gid / In);
    int i = (int)(gid % In);
    size_t rb = (size_t)n * (size_t)In * 9;
    out[rb + i] = f2bf(base[gid]);
    float sc = scaler[gid];
    const float* sp = spline + gid * 8;
    size_t sb = rb + In + (size_t)i * 8;
#pragma unroll
    for (int j = 0; j < 8; ++j) out[sb + j] = f2bf(sp[j] * sc);
}

// ---------------------------------------------------------------------------
// bf16 WMMA GEMM:  C(M,N) = A(M,Kd) * W(N,Kd)^T,  f32 accumulate.
// Block tile 64x256, BK=32, 8 waves (2x4), wave tile 32x64 (2x4 WMMA accs).
// Double-buffered LDS, single barrier per K-stage.
// mode: 0 = store, 1 = gelu+store, 2 = add (residual)
// ---------------------------------------------------------------------------
#define BM 64
#define BN 256
#define BK 32

__global__ __launch_bounds__(256) void kan_gemm_bf16(const unsigned short* __restrict__ A,
                                                     const unsigned short* __restrict__ W,
                                                     float* __restrict__ C,
                                                     int Mdim, int Ndim, int Kd, int mode) {
    __shared__ __align__(16) unsigned short As[2][BM * BK];
    __shared__ __align__(16) unsigned short Bs[2][BN * BK];
    const int tid  = threadIdx.x;
    const int lane = tid & 31;
    const int wave = tid >> 5;
    const int wm = wave >> 2, wn = wave & 3;         // 2 x 4 wave grid
    const int hi = lane >> 4, mr = lane & 15;
    const int blockM = blockIdx.y * BM;
    const int blockN = blockIdx.x * BN;

    v8f acc[2][4];
#pragma unroll
    for (int a = 0; a < 2; ++a)
#pragma unroll
        for (int b = 0; b < 4; ++b)
#pragma unroll
            for (int e = 0; e < 8; ++e) acc[a][b][e] = 0.0f;

    // global staging: A = 1 x 16B/thread, B = 4 x 16B/thread
    const int ar = tid >> 2;                   // 0..63
    const int ac = (tid & 3) * 8;              // 0,8,16,24
    const unsigned short* Ap = A + (size_t)(blockM + ar) * Kd + ac;
    const unsigned short* Bp = W + (size_t)(blockN + tid) * Kd;

    const int nStages = Kd / BK;
    U4 ra, rb0, rb1, rb2, rb3;

    // prologue: stage 0 -> buffer 0
    ra  = *(const U4*)(Ap);
    rb0 = *(const U4*)(Bp + 0);
    rb1 = *(const U4*)(Bp + 8);
    rb2 = *(const U4*)(Bp + 16);
    rb3 = *(const U4*)(Bp + 24);
    *(U4*)&As[0][ar * BK + ac]  = ra;
    *(U4*)&Bs[0][tid * BK + 0]  = rb0;
    *(U4*)&Bs[0][tid * BK + 8]  = rb1;
    *(U4*)&Bs[0][tid * BK + 16] = rb2;
    *(U4*)&Bs[0][tid * BK + 24] = rb3;
    __syncthreads();

    for (int s = 0; s < nStages; ++s) {
        const int p = s & 1;
        if (s + 1 < nStages) {                 // prefetch next stage into regs
            const unsigned short* ap = Ap + (size_t)(s + 1) * BK;
            const unsigned short* bp = Bp + (size_t)(s + 1) * BK;
            ra  = *(const U4*)(ap);
            rb0 = *(const U4*)(bp + 0);
            rb1 = *(const U4*)(bp + 8);
            rb2 = *(const U4*)(bp + 16);
            rb3 = *(const U4*)(bp + 24);
        }

        FragU fa[2], fb[4];
#pragma unroll
        for (int t = 0; t < 2; ++t) {
            int row = wm * 32 + t * 16 + mr;   // A: M = lane&15 per 16-row tile
#pragma unroll
            for (int i = 0; i < 16; ++i) {     // K octet selected by lane-half
                int k = ((i < 8) ? i : (i + 8)) + hi * 8;
                fa[t].u[i] = As[p][row * BK + k];
            }
        }
#pragma unroll
        for (int t = 0; t < 4; ++t) {
            int col = wn * 64 + t * 16 + mr;   // B: N = lane&15
#pragma unroll
            for (int i = 0; i < 16; ++i) {
                int k = ((i < 8) ? i : (i + 8)) + hi * 8; // lanes 0-15: K 0..7,16..23; 16-31: 8..15,24..31
                fb[t].u[i] = Bs[p][col * BK + k];
            }
        }
#pragma unroll
        for (int tm = 0; tm < 2; ++tm)
#pragma unroll
            for (int tn = 0; tn < 4; ++tn)
                acc[tm][tn] = __builtin_amdgcn_wmma_f32_16x16x32_bf16(
                    false, fa[tm].b, false, fb[tn].b,
                    (short)0, acc[tm][tn], false, false);

        if (s + 1 < nStages) {                 // stage regs -> alternate buffer
            const int q = p ^ 1;
            *(U4*)&As[q][ar * BK + ac]  = ra;
            *(U4*)&Bs[q][tid * BK + 0]  = rb0;
            *(U4*)&Bs[q][tid * BK + 8]  = rb1;
            *(U4*)&Bs[q][tid * BK + 16] = rb2;
            *(U4*)&Bs[q][tid * BK + 24] = rb3;
            __syncthreads();
        }
    }

#pragma unroll
    for (int tm = 0; tm < 2; ++tm) {
#pragma unroll
        for (int tn = 0; tn < 4; ++tn) {
            int col = blockN + wn * 64 + tn * 16 + mr;
#pragma unroll
            for (int r = 0; r < 8; ++r) {
                int row = blockM + wm * 32 + tm * 16 + r + hi * 8; // C: M = r + 8*hi
                size_t idx = (size_t)row * (size_t)Ndim + col;
                float v = acc[tm][tn][r];
                if (mode == 1) {
                    float u = v + 0.044715f * v * v * v;
                    v = 0.5f * v * (1.0f + tanhf(0.7978845608028654f * u));
                }
                if (mode == 2) C[idx] += v;
                else           C[idx] = v;
            }
        }
    }
}

// ---------------------------------------------------------------------------
// Fused causal attention: one block per (q, head, batch). hd=64, T=1024.
// qkv layout: (B*T, 1536) = [q | k | v], head h at columns h*64.
// ---------------------------------------------------------------------------
__global__ __launch_bounds__(256) void attn_kernel(const float* __restrict__ qkv,
                                                   float* __restrict__ Y) {
    constexpr int T = 1024, HD = 64, E3 = 1536;
    __shared__ float qv[HD];
    __shared__ float sc[T];
    __shared__ float red[256];
    int q = blockIdx.x, h = blockIdx.y, b = blockIdx.z, tid = threadIdx.x;

    const float* qr = qkv + ((size_t)(b * T + q)) * E3 + h * HD;
    if (tid < HD) qv[tid] = qr[tid];
    __syncthreads();

    for (int k = tid; k < T; k += 256) {
        float s = -3.0e38f;
        if (k <= q) {
            const float* kr = qkv + ((size_t)(b * T + k)) * E3 + 512 + h * HD;
            float d = 0.0f;
#pragma unroll 8
            for (int e = 0; e < HD; ++e) d += qv[e] * kr[e];
            s = d * 0.125f;                      // 1/sqrt(64)
        }
        sc[k] = s;
    }
    __syncthreads();

    float mloc = -3.0e38f;
    for (int k = tid; k < T; k += 256) mloc = fmaxf(mloc, sc[k]);
    red[tid] = mloc; __syncthreads();
    for (int s = 128; s > 0; s >>= 1) {
        if (tid < s) red[tid] = fmaxf(red[tid], red[tid + s]);
        __syncthreads();
    }
    float mx = red[0]; __syncthreads();

    float ploc = 0.0f;
    for (int k = tid; k < T; k += 256) {
        float p = (k <= q) ? __expf(sc[k] - mx) : 0.0f;
        sc[k] = p;
        ploc += p;
    }
    red[tid] = ploc; __syncthreads();
    for (int s = 128; s > 0; s >>= 1) {
        if (tid < s) red[tid] += red[tid + s];
        __syncthreads();
    }
    float S = red[0]; __syncthreads();

    int d = tid & 63, part = tid >> 6;
    float acc = 0.0f;
    for (int k = part; k <= q; k += 4) {
        const float* vr = qkv + ((size_t)(b * T + k)) * E3 + 1024 + h * HD;
        acc += sc[k] * vr[d];
    }
    red[tid] = acc; __syncthreads();
    if (part == 0) {
        float tot = red[d] + red[d + 64] + red[d + 128] + red[d + 192];
        Y[((size_t)(b * T + q)) * 512 + h * HD + d] = tot / S;
    }
}

// ---------------------------------------------------------------------------
// Host orchestration
// ---------------------------------------------------------------------------
extern "C" void kernel_launch(void* const* d_in, const int* in_sizes, int n_in,
                              void* d_out, int out_size, void* d_ws, size_t ws_size,
                              hipStream_t stream) {
    (void)in_sizes; (void)n_in; (void)out_size; (void)ws_size;
    constexpr int Bb = 2, T = 1024, E = 512, V = 32000, L = 2, NH = 8;
    constexpr int M = Bb * T;                    // 2048

    const int*   idx       = (const int*)  d_in[0];
    const float* wte       = (const float*)d_in[1];
    const float* wpe       = (const float*)d_in[2];
    const float* ln1_w     = (const float*)d_in[3];
    const float* ln1_b     = (const float*)d_in[4];
    const float* ln2_w     = (const float*)d_in[5];
    const float* ln2_b     = (const float*)d_in[6];
    const float* ca_base   = (const float*)d_in[7];
    const float* ca_spline = (const float*)d_in[8];
    const float* ca_scaler = (const float*)d_in[9];
    const float* cp_base   = (const float*)d_in[10];
    const float* cp_spline = (const float*)d_in[11];
    const float* cp_scaler = (const float*)d_in[12];
    const float* fc_base   = (const float*)d_in[13];
    const float* fc_spline = (const float*)d_in[14];
    const float* fc_scaler = (const float*)d_in[15];
    const float* pr_base   = (const float*)d_in[16];
    const float* pr_spline = (const float*)d_in[17];
    const float* pr_scaler = (const float*)d_in[18];
    const float* lnf_w     = (const float*)d_in[19];
    const float* lnf_b     = (const float*)d_in[20];
    const float* hd_base   = (const float*)d_in[21];
    const float* hd_spline = (const float*)d_in[22];
    const float* hd_scaler = (const float*)d_in[23];

    // workspace carve (256B aligned)
    char* wsp = (char*)d_ws;
    size_t off = 0;
    auto take = [&](size_t bytes) -> void* {
        off = (off + 255) & ~(size_t)255;
        void* p = (void*)(wsp + off);
        off += bytes;
        return p;
    };
    float* x    = (float*)take((size_t)M * E * 4);
    float* h    = (float*)take((size_t)M * E * 4);
    float* y    = (float*)take((size_t)M * E * 4);
    float* qkv  = (float*)take((size_t)M * 3 * E * 4);
    float* mbuf = (float*)take((size_t)M * 4 * E * 4);
    unsigned short* Aact = (unsigned short*)take((size_t)M * (4 * E * 9) * 2);   // up to Kd=18432
    unsigned short* Wb   = (unsigned short*)take((size_t)V * (E * 9) * 2);       // up to 32000x4608

    auto gridFor = [](size_t total) { return (unsigned)((total + 255) / 256); };

    embed_kernel<<<M, 256, 0, stream>>>(idx, wte, wpe, x);

    for (int l = 0; l < L; ++l) {
        // ---- attention branch: qkv = kan(ln1(x), ca)
        ln_kernel<<<M, 256, 0, stream>>>(x, ln1_w + (size_t)l * E, ln1_b + (size_t)l * E, h);
        kan_prep_act<<<gridFor((size_t)M * E), 256, 0, stream>>>(Aact, h, M, E);
        kan_prep_w<<<gridFor((size_t)3 * E * E), 256, 0, stream>>>(
            Wb, ca_base + (size_t)l * 3 * E * E, ca_spline + (size_t)l * 3 * E * E * 8,
            ca_scaler + (size_t)l * 3 * E * E, 3 * E, E);
        kan_gemm_bf16<<<dim3(3 * E / BN, M / BM), 256, 0, stream>>>(
            Aact, Wb, qkv, M, 3 * E, E * 9, 0);

        attn_kernel<<<dim3(T, NH, Bb), 256, 0, stream>>>(qkv, y);

        // ---- x += kan(y, cp)
        kan_prep_act<<<gridFor((size_t)M * E), 256, 0, stream>>>(Aact, y, M, E);
        kan_prep_w<<<gridFor((size_t)E * E), 256, 0, stream>>>(
            Wb, cp_base + (size_t)l * E * E, cp_spline + (size_t)l * E * E * 8,
            cp_scaler + (size_t)l * E * E, E, E);
        kan_gemm_bf16<<<dim3(E / BN, M / BM), 256, 0, stream>>>(
            Aact, Wb, x, M, E, E * 9, 2);

        // ---- m = gelu(kan(ln2(x), fc))
        ln_kernel<<<M, 256, 0, stream>>>(x, ln2_w + (size_t)l * E, ln2_b + (size_t)l * E, h);
        kan_prep_act<<<gridFor((size_t)M * E), 256, 0, stream>>>(Aact, h, M, E);
        kan_prep_w<<<gridFor((size_t)4 * E * E), 256, 0, stream>>>(
            Wb, fc_base + (size_t)l * 4 * E * E, fc_spline + (size_t)l * 4 * E * E * 8,
            fc_scaler + (size_t)l * 4 * E * E, 4 * E, E);
        kan_gemm_bf16<<<dim3(4 * E / BN, M / BM), 256, 0, stream>>>(
            Aact, Wb, mbuf, M, 4 * E, E * 9, 1);

        // ---- x += kan(m, pr)   (Kd = 4E*9 = 18432)
        kan_prep_act<<<gridFor((size_t)M * 4 * E), 256, 0, stream>>>(Aact, mbuf, M, 4 * E);
        kan_prep_w<<<gridFor((size_t)E * 4 * E), 256, 0, stream>>>(
            Wb, pr_base + (size_t)l * E * 4 * E, pr_spline + (size_t)l * E * 4 * E * 8,
            pr_scaler + (size_t)l * E * 4 * E, E, 4 * E);
        kan_gemm_bf16<<<dim3(E / BN, M / BM), 256, 0, stream>>>(
            Aact, Wb, x, M, E, 4 * E * 9, 2);
    }

    // ---- logits = kan(lnf(x), head) -> d_out
    ln_kernel<<<M, 256, 0, stream>>>(x, lnf_w, lnf_b, h);
    kan_prep_act<<<gridFor((size_t)M * E), 256, 0, stream>>>(Aact, h, M, E);
    kan_prep_w<<<gridFor((size_t)V * E), 256, 0, stream>>>(
        Wb, hd_base, hd_spline, hd_scaler, V, E);
    kan_gemm_bf16<<<dim3(V / BN, M / BM), 256, 0, stream>>>(
        Aact, Wb, (float*)d_out, M, V, E * 9, 0);
}
